// Net_1778116460596
// MI455X (gfx1250) — compile-verified, matrix operand order checked by reading, not run on hardware
//
#include <hip/hip_runtime.h>

typedef __attribute__((ext_vector_type(16))) _Float16 v16h;
typedef __attribute__((ext_vector_type(2)))  _Float16 h2;
typedef __attribute__((ext_vector_type(8)))  float    v8f;

#define ROWS 4096      // B*N
#define DD   128       // hidden dim
#define NJ   64        // message/softmax width
#define OUTW 277       // 20 logits + 1 value + 128 H_E + 128 H_I

__device__ __forceinline__ float sigmoidf_(float x){ return 1.f/(1.f+__expf(-x)); }
__device__ __forceinline__ float clamp16_(float v){ return fminf(fmaxf(v,-65504.f),65504.f); }

// ---------------------------------------------------------------------------
// Pack a weight matrix W[K,N] (fp32 row-major) into f16 "fragment order":
//   out[(((t*KC)+kc)*32 + lane)*16 + e] = W[kk, t*16 + (lane&15)]  (0 if OOB)
//   kk = kc*32 + ((v>>2)<<4) + ((lane>>4)<<3) + ((v&3)<<1) + (e&1),  v = e>>1
// so a wave's B fragment is one contiguous 1 KB chunk (32 B per lane) per (tile, kchunk).
// ---------------------------------------------------------------------------
__global__ void pack_w(const float* __restrict__ W, int K, int N, int KC,
                       _Float16* __restrict__ out, int total)
{
  int idx = blockIdx.x*blockDim.x + threadIdx.x;
  if (idx >= total) return;
  int e   = idx & 15;
  int l   = (idx >> 4) & 31;
  int blk = idx >> 9;            // t*KC + kc
  int kc  = blk % KC;
  int t   = blk / KC;
  int v   = e >> 1;
  int kk  = kc*32 + ((v>>2)<<4) + ((l>>4)<<3) + ((v&3)<<1) + (e&1);
  int n   = t*16 + (l & 15);
  float val = (n < N) ? W[kk*N + n] : 0.f;
  out[idx] = (_Float16)clamp16_(val);
}

// ---------------------------------------------------------------------------
// Generic WMMA GEMM:  C[r, col0+n] = act( A[r,0:K] @ W[K,N] + bias[n] )
// One wave per 16x16 tile. A staged fp32->f16 into LDS in fragment order
// (float2 loads, half2 stores); B from pre-packed fragment buffer.
// Inner loop: 2x ds_load_b128 + 2x global_load_b128 + v_wmma. K % 32 == 0.
// ---------------------------------------------------------------------------
__global__ void gemm_bias_act(const float* __restrict__ A, int K,
                              const _Float16* __restrict__ Wp, int N,
                              const float* __restrict__ bias,
                              float* __restrict__ C, int ldc, int col0,
                              int act, int nanfix)
{
  __shared__ __align__(16) _Float16 Alds[16*288];   // fragment-ordered, max K=288
  const int bx = blockIdx.x, by = blockIdx.y;
  const int lane = threadIdx.x;
  const int KC = K >> 5;

  // stage A tile: one half2 (two consecutive K values) per iteration
  const int pairs = K << 3;                         // KC * 256
  for (int idx = lane; idx < pairs; idx += 32){
    int kc   = idx >> 8;
    int rem  = idx & 255;
    int l    = rem >> 3, v = rem & 7;
    int kk   = kc*32 + ((v>>2)<<4) + ((l>>4)<<3) + ((v&3)<<1);
    int m    = l & 15;
    float2 p = *(const float2*)(A + (bx*16 + m)*K + kk);
    float x0 = p.x, x1 = p.y;
    if (nanfix){ if (!(x0==x0)) x0 = 0.f; if (!(x1==x1)) x1 = 0.f; }
    h2 q; q.x = (_Float16)clamp16_(x0); q.y = (_Float16)clamp16_(x1);
    *(h2*)&Alds[((kc*32 + l)<<4) + (v<<1)] = q;
  }
  __syncthreads();

  const int half = lane >> 4;
  const int l15  = lane & 15;
  const int n    = by*16 + l15;

  v8f acc;
  #pragma unroll
  for (int i=0;i<8;++i) acc[i] = 0.f;

  const _Float16* wptr = Wp + (((by*KC)*32 + lane) << 4);
  for (int kc = 0; kc < KC; ++kc){
    v16h af = *(const v16h*)&Alds[((kc*32 + lane)<<4)];
    v16h bf = *(const v16h*)(wptr + (kc << 9));     // kc*32*16
    acc = __builtin_amdgcn_wmma_f32_16x16x32_f16(false, af, false, bf,
                                                 (short)0, acc, false, false);
  }

  if (n < N){
    float b = bias ? bias[n] : 0.f;
    #pragma unroll
    for (int e=0; e<8; ++e){
      int m = e + half*8;                           // C/D layout: VGPR e -> row
      float o = acc[e] + b;
      if (act == 1) o = fmaxf(o, 0.f);
      C[(bx*16 + m)*ldc + col0 + n] = o;
    }
  }
}

// ---------------------------------------------------------------------------
// torch GRUCell, zero hidden state. gi = [ROWS,384] already includes bih.
// ---------------------------------------------------------------------------
__global__ void gru_zero_state(const float* __restrict__ gi,
                               const float* __restrict__ bhh,
                               float* __restrict__ g)
{
  int idx = blockIdx.x*blockDim.x + threadIdx.x;
  if (idx >= ROWS*DD) return;
  int row = idx >> 7, d = idx & 127;
  const float* gr = gi + row*384;
  float r  = sigmoidf_(gr[d]       + bhh[d]);
  float z  = sigmoidf_(gr[128 + d] + bhh[128 + d]);
  float nn = tanhf   (gr[256 + d] + r*bhh[256 + d]);
  g[idx] = (1.f - z)*nn;
}

// s[r] = hw[r,:] . a1
__global__ void rowdot(const float* __restrict__ X, const float* __restrict__ a,
                       float* __restrict__ s)
{
  int r = blockIdx.x*blockDim.x + threadIdx.x;
  if (r >= ROWS) return;
  float acc = 0.f;
  #pragma unroll 8
  for (int d=0; d<DD; ++d) acc += X[r*DD + d]*a[d];
  s[r] = acc;
}

// ---------------------------------------------------------------------------
// Fused E-GAT, one workgroup (256 thr = 8 waves) per (b,i):
//  Hm = message[b,i] @ W (WMMA into LDS; message read from HBM exactly once),
//  masked softmax attention, alpha-weighted sum, elu((hw+ws)@Wo+bo) -> out.
// Wp is the fragment-packed g_W (8 tiles x 4 kchunks).
// ---------------------------------------------------------------------------
__global__ __launch_bounds__(256)
void gat_fused(const float* __restrict__ message,   // [ROWS, NJ, DD]
               const _Float16* __restrict__ Wp,     // packed [8*4*32*16]
               const float* __restrict__ a2,        // [DD]
               const float* __restrict__ s,         // [ROWS]
               const float* __restrict__ hw,        // [ROWS, DD]
               const float* __restrict__ Wo,        // [DD, DD]
               const float* __restrict__ bo,        // [DD]
               const int*   __restrict__ mask,      // [ROWS, NJ]
               float* __restrict__ out, int ldc, int col0)
{
  __shared__ __align__(16) _Float16 Apack[NJ*DD];   // 16 KB, fragment-ordered
  __shared__ float Hm[NJ*132];                      // ~33 KB, padded stride
  __shared__ float a2s[DD];
  __shared__ float em[NJ];
  __shared__ float al[NJ];
  __shared__ float hws[DD];
  __shared__ float red[2];

  const int row = blockIdx.x;
  const int tid = threadIdx.x;
  const float* msg = message + row*(NJ*DD);

  // stage message slice in fragment order: Apack[((mt*4+kc)*32 + l)*16 + 2v]
  for (int idx = tid; idx < (NJ*DD)/2; idx += 256){ // 4096 half2 pairs
    int mt   = idx >> 10;
    int rem  = idx & 1023;
    int kc   = rem >> 8;
    int rem2 = rem & 255;
    int l    = rem2 >> 3, v = rem2 & 7;
    int kk   = kc*32 + ((v>>2)<<4) + ((l>>4)<<3) + ((v&3)<<1);
    int m    = mt*16 + (l & 15);
    float2 p = *(const float2*)(msg + m*DD + kk);
    h2 q; q.x = (_Float16)clamp16_(p.x); q.y = (_Float16)clamp16_(p.y);
    *(h2*)&Apack[(((mt*4 + kc)*32 + l)<<4) + (v<<1)] = q;
  }
  if (tid < DD) a2s[tid] = a2[tid];
  __syncthreads();

  // Hm = msg @ W : wave wv owns column tile wv; B fragments reused over 4 M-tiles
  const int wv   = tid >> 5;
  const int lane = tid & 31;
  const int half = lane >> 4;
  const int l15  = lane & 15;
  const int n    = wv*16 + l15;

  v16h bfr[4];
  #pragma unroll
  for (int kc=0; kc<4; ++kc)
    bfr[kc] = *(const v16h*)(Wp + (((wv*4 + kc)*32 + lane) << 4));

  #pragma unroll
  for (int mt=0; mt<4; ++mt){
    v8f acc;
    #pragma unroll
    for (int i=0;i<8;++i) acc[i] = 0.f;
    #pragma unroll
    for (int kc=0; kc<4; ++kc){
      v16h af = *(const v16h*)&Apack[(((mt*4 + kc)*32 + lane)<<4)];
      acc = __builtin_amdgcn_wmma_f32_16x16x32_f16(false, af, false, bfr[kc],
                                                   (short)0, acc, false, false);
    }
    #pragma unroll
    for (int e=0; e<8; ++e){
      int m = mt*16 + e + half*8;
      Hm[m*132 + n] = acc[e];
    }
  }
  __syncthreads();

  // attention logits (leaky_relu 0.2, multiplicative mask BEFORE softmax)
  const float srow = s[row];
  if (tid < NJ){
    float t = 0.f;
    for (int d=0; d<DD; ++d) t += Hm[tid*132 + d]*a2s[d];
    float e = srow + t;
    e = (e > 0.f) ? e : 0.2f*e;
    em[tid] = e * (float)mask[row*NJ + tid];
  }
  __syncthreads();
  if (tid == 0){
    float mx = -3.0e38f;
    for (int j=0;j<NJ;++j) mx = fmaxf(mx, em[j]);
    red[0] = mx;
  }
  __syncthreads();
  if (tid < NJ) em[tid] = __expf(em[tid] - red[0]);
  __syncthreads();
  if (tid == 0){
    float sm = 0.f;
    for (int j=0;j<NJ;++j) sm += em[j];
    red[1] = sm;
  }
  __syncthreads();
  if (tid < NJ){
    float a = em[tid]/red[1] * (float)mask[row*NJ + tid];
    if (!((a - a) == 0.f)) a = 0.f;                 // nan_to_num
    al[tid] = a;
  }
  __syncthreads();

  // alpha-weighted sum over messages + residual
  if (tid < DD){
    float acc = 0.f;
    for (int j=0;j<NJ;++j) acc += al[j]*Hm[j*132 + tid];
    hws[tid] = hw[row*DD + tid] + acc;
  }
  __syncthreads();

  // output projection + ELU
  if (tid < DD){
    float o = bo[tid];
    for (int d=0; d<DD; ++d) o += hws[d]*Wo[d*DD + tid];
    o = (o > 0.f) ? o : (__expf(o) - 1.f);
    out[row*ldc + col0 + tid] = o;
  }
}

// ---------------------------------------------------------------------------
extern "C" void kernel_launch(void* const* d_in, const int* in_sizes, int n_in,
                              void* d_out, int out_size, void* d_ws, size_t ws_size,
                              hipStream_t stream)
{
  (void)in_sizes; (void)n_in; (void)out_size; (void)ws_size;

  const float* obs     = (const float*)d_in[0];   // [4096, 288]
  const float* act_in  = (const float*)d_in[1];   // [4096, 32]
  const float* msg_obs = (const float*)d_in[2];   // [4096, 64, 128]
  const float* msg_adv = (const float*)d_in[3];
  const int*   mask    = (const int*)d_in[4];     // [4096, 64]
  const float* P[44];
  for (int i=0;i<44;++i) P[i] = (const float*)d_in[5+i];
  const float *enc_w1=P[0], *enc_b1=P[1], *enc_w2=P[2], *enc_b2=P[3];
  const float *abs_w1=P[4], *abs_b1=P[5], *abs_w2=P[6], *abs_b2=P[7];
  const float *aab_w1=P[8], *aab_b1=P[9], *aab_w2=P[10], *aab_b2=P[11];
  const float *fc1_w=P[12], *fc1_b=P[13];
  const float *gru_wih=P[14], *gru_bih=P[16], *gru_bhh=P[17];  // gru_whh unused
  const float *fc2_w=P[18], *fc2_b=P[19];
  const float *g1_W=P[20], *g1_a1=P[21], *g1_a2=P[22], *g1_Wo=P[23], *g1_bo=P[24];
  const float *g2_W=P[25], *g2_a1=P[26], *g2_a2=P[27], *g2_Wo=P[28], *g2_bo=P[29];
  const float *pg_w1=P[30], *pg_b1=P[31], *pg_w2=P[32], *pg_b2=P[33];
  const float *pol_w1=P[34], *pol_b1=P[35], *pol_w2=P[36], *pol_b2=P[37],
              *pol_w3=P[38], *pol_b3=P[39];
  const float *val_w1=P[40], *val_b1=P[41], *val_w2=P[42], *val_b2=P[43];

  float* out = (float*)d_out;
  float* ws  = (float*)d_ws;

  // workspace layout (floats); lifetimes overlapped
  float* H1     = ws;                 // [4096,128] scratch hidden
  float* OBSENC = ws + 524288;        // [4096,128]
  float* AB     = ws + 1048576;       // [4096,256] concat(ab_o, ab_a)
  float* GRUIN  = ws + 2097152;       // [4096,128]; later reused as H2P [4096,64]
  float* GI     = ws + 2621440;       // [4096,384]; later HW1/HW2/S1/S2
  float* G      = ws + 4194304;       // [4096,128]
  float* HADV   = ws + 4718592;       // [4096,128]
  float* HSUM   = ws + 5242880;       // [4096,128]
  float* H2P    = GRUIN;
  float* HW1    = ws + 2621440;
  float* HW2    = ws + 3145728;
  float* S1     = ws + 3670016;
  float* S2     = ws + 3674112;
  _Float16* wp_base = (_Float16*)(ws + 5767168);  // packed f16 weights (~0.67 MB)

  size_t cur = 0;
  auto pack = [&](const float* Wm, int K, int N) -> const _Float16* {
    int KC = K >> 5;
    int Nt = (N + 15) >> 4;
    int total = Nt * KC * 512;
    _Float16* dst = wp_base + cur;
    cur += (size_t)total;
    pack_w<<<dim3((total + 255)/256), dim3(256), 0, stream>>>(Wm, K, N, KC, dst, total);
    return dst;
  };

  const _Float16* enc_w1p = pack(enc_w1, 288, 128);
  const _Float16* enc_w2p = pack(enc_w2, 128, 128);
  const _Float16* abs_w1p = pack(abs_w1, 288, 128);
  const _Float16* abs_w2p = pack(abs_w2, 128, 128);
  const _Float16* aab_w1p = pack(aab_w1,  32, 128);
  const _Float16* aab_w2p = pack(aab_w2, 128, 128);
  const _Float16* fc1_wp  = pack(fc1_w,  256, 128);
  const _Float16* gru_wp  = pack(gru_wih,128, 384);
  const _Float16* fc2_wp  = pack(fc2_w,  128, 128);
  const _Float16* pg_w1p  = pack(pg_w1,  128, 128);
  const _Float16* pg_w2p  = pack(pg_w2,  128, 128);
  const _Float16* pol_w1p = pack(pol_w1, 128, 128);
  const _Float16* pol_w2p = pack(pol_w2, 128,  64);
  const _Float16* pol_w3p = pack(pol_w3,  64,  20);
  const _Float16* val_w1p = pack(val_w1, 128, 128);
  const _Float16* val_w2p = pack(val_w2, 128,   1);
  const _Float16* g1_Wp   = pack(g1_W,   128, 128);
  const _Float16* g2_Wp   = pack(g2_W,   128, 128);

  auto gemm = [&](const float* A, int K, const _Float16* Wp, int N, const float* b,
                  float* C, int ldc, int col0, int act, int nanfix){
    dim3 grid(ROWS/16, (N + 15)/16);
    gemm_bias_act<<<grid, dim3(32), 0, stream>>>(A, K, Wp, N, b, C, ldc, col0, act, nanfix);
  };

  // ---- row MLP pipeline -------------------------------------------------
  gemm(obs,    288, enc_w1p, 128, enc_b1, H1,     128,   0, 1, 1);  // enc hidden (nan_to_num)
  gemm(H1,     128, enc_w2p, 128, enc_b2, OBSENC, 128,   0, 0, 0);  // obs_enc
  gemm(obs,    288, abs_w1p, 128, abs_b1, H1,     128,   0, 1, 1);  // abs hidden
  gemm(H1,     128, abs_w2p, 128, abs_b2, AB,     256,   0, 0, 0);  // ab_o -> AB[:,0:128]
  gemm(act_in,  32, aab_w1p, 128, aab_b1, H1,     128,   0, 1, 0);  // aab hidden
  gemm(H1,     128, aab_w2p, 128, aab_b2, AB,     256, 128, 0, 0);  // ab_a -> AB[:,128:256]
  gemm(AB,     256, fc1_wp,  128, fc1_b,  GRUIN,  128,   0, 1, 0);  // gru_in
  gemm(GRUIN,  128, gru_wp,  384, gru_bih,GI,     384,   0, 0, 0);  // gi = x@wih + bih
  gru_zero_state<<<dim3((ROWS*DD + 255)/256), dim3(256), 0, stream>>>(GI, gru_bhh, G);
  gemm(G,      128, fc2_wp,  128, fc2_b,  HADV,   128,   0, 0, 0);  // h_adv
  gemm(HADV,   128, pg_w1p,  128, pg_b1,  H1,     128,   0, 1, 0);
  gemm(H1,     128, pg_w2p,  128, pg_b2,  HSUM,   128,   0, 0, 0);  // H_sum
  gemm(HSUM,   128, pol_w1p, 128, pol_b1, H1,     128,   0, 1, 0);  // policy h1
  gemm(H1,     128, pol_w2p,  64, pol_b2, H2P,     64,   0, 1, 0);  // policy h2
  gemm(H2P,     64, pol_w3p,  20, pol_b3, out,   OUTW,   0, 0, 0);  // logits -> out[:,0:20]
  gemm(HSUM,   128, val_w1p, 128, val_b1, H1,     128,   0, 1, 0);  // value hidden
  gemm(H1,     128, val_w2p,   1, val_b2, out,   OUTW,  20, 0, 0);  // value  -> out[:,20]

  // ---- E-GAT 1 (H_E) ----------------------------------------------------
  gemm(OBSENC, 128, g1_Wp,   128, nullptr, HW1,   128,   0, 0, 0);  // hw = h@W
  rowdot<<<dim3((ROWS + 127)/128), dim3(128), 0, stream>>>(HW1, g1_a1, S1);
  gat_fused<<<dim3(ROWS), dim3(256), 0, stream>>>(msg_obs, g1_Wp, g1_a2, S1, HW1,
                                                  g1_Wo, g1_bo, mask, out, OUTW, 21);

  // ---- E-GAT 2 (H_I) ----------------------------------------------------
  gemm(HADV,   128, g2_Wp,   128, nullptr, HW2,   128,   0, 0, 0);
  rowdot<<<dim3((ROWS + 127)/128), dim3(128), 0, stream>>>(HW2, g2_a1, S2);
  gat_fused<<<dim3(ROWS), dim3(256), 0, stream>>>(msg_adv, g2_Wp, g2_a2, S2, HW2,
                                                  g2_Wo, g2_bo, mask, out, OUTW, 149);
}
